// MultiHeadAttention_19086834663925
// MI455X (gfx1250) — matrix operand order repaired and checked by reading, not compiled, not run on hardware
//
#include <hip/hip_runtime.h>
#include <math.h>

typedef __attribute__((ext_vector_type(16))) __bf16      v16bf;
typedef __attribute__((ext_vector_type(8)))  float       v8f;
typedef __attribute__((ext_vector_type(4)))  unsigned int v4u;
typedef unsigned short u16;

#define D_MODEL 2048
#define SEQ     2048
#define BATCH   2
#define HEADS   16
#define DKH     128              // d_k per head
#define MTOT    (BATCH * SEQ)    // 4096 rows

// 32-byte fragment: 16 bf16 values, viewed as two 16B chunks for LDS/global loads
union FragAB {
    v16bf v;
    v4u   q[2];
    u16   h[16];
};

__device__ __forceinline__ u16 f2bf(float f) {
    unsigned u = __float_as_uint(f);
    return (u16)((u + 0x7FFFu + ((u >> 16) & 1u)) >> 16);   // round-nearest-even
}

// ---------------------------------------------------------------------------
// CDNA5 async copy: 16B per lane, global -> LDS, tracked by ASYNCcnt.
// LDS dest passed as low 32 bits of the generic pointer (== LDS byte offset).
// ---------------------------------------------------------------------------
__device__ __forceinline__ void async_copy16(unsigned lds_addr, const void* gaddr) {
    asm volatile("global_load_async_to_lds_b128 %0, %1, off"
                 :: "v"(lds_addr), "v"(gaddr)
                 : "memory");
}
#define S_WAIT_ASYNC_0 asm volatile("s_wait_asynccnt 0x0" ::: "memory")
#define S_WAIT_ASYNC_4 asm volatile("s_wait_asynccnt 0x4" ::: "memory")
__device__ __forceinline__ unsigned lds_off(const void* p) {
    return (unsigned)(size_t)p;
}

// ---------------------------------------------------------------------------
// f32 -> bf16 conversion (grid-stride)
// ---------------------------------------------------------------------------
__global__ void cvt_f32_bf16(const float* __restrict__ in, u16* __restrict__ out, int n) {
    int i      = blockIdx.x * blockDim.x + threadIdx.x;
    int stride = gridDim.x * blockDim.x;
    for (; i < n; i += stride) out[i] = f2bf(in[i]);
}

// ---------------------------------------------------------------------------
// Y[M,N] = A[M,K] @ W[N,K]^T   (bf16 inputs, f32 accumulate via WMMA)
// Block: 256 threads (8 waves), 128x128 tile. Wave: 32x64 (2x4 WMMA tiles).
// Double-buffered LDS fed by global_load_async_to_lds_b128.
// ---------------------------------------------------------------------------
template <bool BF16OUT>
__global__ __launch_bounds__(256) void gemm_xwT(const u16* __restrict__ A,
                                                const u16* __restrict__ W,
                                                void* __restrict__ out,
                                                int M, int N, int K, float scale) {
    __shared__ u16 As[2][128][40];   // 128 x 32 K-slice, pad 8 (row stride 80B)
    __shared__ u16 Bs[2][128][40];   // 128 N-rows of W x 32 K-slice

    const int tid  = threadIdx.x;
    const int lane = tid & 31;
    const int wid  = tid >> 5;          // 0..7
    const int half = (lane >> 4) & 1;   // K-half / M-half selector per ISA layout
    const int r    = lane & 15;
    const int wm   = wid & 3;           // wave M offset *32
    const int wn   = wid >> 2;          // wave N offset *64

    const int mblock = blockIdx.y * 128;
    const int nblock = blockIdx.x * 128;

    v8f zero = {0.f, 0.f, 0.f, 0.f, 0.f, 0.f, 0.f, 0.f};
    v8f acc[2][4];
    for (int mi = 0; mi < 2; ++mi)
        for (int ni = 0; ni < 4; ++ni) acc[mi][ni] = zero;

    const int ldrow   = tid >> 1;         // 0..127
    const int ldchunk = (tid & 1) * 16;   // 0 or 16 (elements)

    // issue one K-slice (4 x 16B per thread) asynchronously into buffer `buf`
    auto issue_tile = [&](int buf, int k0) {
        const u16* ga = &A[(size_t)(mblock + ldrow) * K + k0 + ldchunk];
        async_copy16(lds_off(&As[buf][ldrow][ldchunk]),     ga);
        async_copy16(lds_off(&As[buf][ldrow][ldchunk + 8]), ga + 8);
        const u16* gw = &W[(size_t)(nblock + ldrow) * K + k0 + ldchunk];
        async_copy16(lds_off(&Bs[buf][ldrow][ldchunk]),     gw);
        async_copy16(lds_off(&Bs[buf][ldrow][ldchunk + 8]), gw + 8);
    };

    const int nt = K / 32;
    issue_tile(0, 0);
    for (int t = 0; t < nt; ++t) {
        const int cur = t & 1;
        if (t + 1 < nt) {
            issue_tile(cur ^ 1, (t + 1) * 32);
            S_WAIT_ASYNC_4;   // newest 4 (next tile) may remain in flight
        } else {
            S_WAIT_ASYNC_0;
        }
        __syncthreads();      // all waves' async writes to buffer `cur` visible

        // --- build fragments per ISA VGPR layouts ---
        FragAB af[2], bfr[4];
        for (int mi = 0; mi < 2; ++mi) {
            int row     = wm * 32 + mi * 16 + r;                       // A: lane r = row M
            af[mi].q[0] = *(const v4u*)&As[cur][row][half * 8];        // K = half*8 .. +7
            af[mi].q[1] = *(const v4u*)&As[cur][row][16 + half * 8];   // K = 16+half*8 ..
        }
        for (int ni = 0; ni < 4; ++ni) {
            int row      = wn * 64 + ni * 16 + r;                      // B: lane r = col N
            bfr[ni].q[0] = *(const v4u*)&Bs[cur][row][half * 8];       // K = half*8 .. +7
            bfr[ni].q[1] = *(const v4u*)&Bs[cur][row][16 + half * 8];
        }
        for (int mi = 0; mi < 2; ++mi)
            for (int ni = 0; ni < 4; ++ni)
                acc[mi][ni] = __builtin_amdgcn_wmma_f32_16x16x32_bf16(
                    false, af[mi].v, false, bfr[ni].v, (short)0, acc[mi][ni], false, false);
        __syncthreads();      // done reading `cur` before it is refilled
    }

    // --- store: C layout row M = half*8 + i, col N = r ---
    for (int mi = 0; mi < 2; ++mi)
        for (int ni = 0; ni < 4; ++ni)
            for (int i = 0; i < 8; ++i) {
                int    row = mblock + wm * 32 + mi * 16 + half * 8 + i;
                int    col = nblock + wn * 64 + ni * 16 + r;
                float  v   = acc[mi][ni][i] * scale;
                if (BF16OUT) ((u16*)out)[(size_t)row * N + col] = f2bf(v);
                else         ((float*)out)[(size_t)row * N + col] = v;
            }
}

// ---------------------------------------------------------------------------
// Causal flash attention, bf16 WMMA, f32 online softmax.
// Block: 256 threads = 8 waves; wave owns 16 q-rows; 32-key tiles.
// K tile streamed via async-to-LDS; V tile transposed in LDS.
// Q is pre-scaled by 1/sqrt(d_k).
// ---------------------------------------------------------------------------
#define KT 32

__global__ __launch_bounds__(256) void flash_attn(const u16* __restrict__ Q,
                                                  const u16* __restrict__ Km,
                                                  const u16* __restrict__ Vm,
                                                  u16* __restrict__ O) {
    __shared__ u16 Ks[KT][DKH + 8];    // keys x d_k (row-major)
    __shared__ u16 Vt[DKH][KT + 8];    // transposed: d x keys  (PV B-frags contiguous)
    __shared__ u16 Ps[8][16][KT + 8];  // per-wave probabilities (bf16)

    const int tid  = threadIdx.x;
    const int lane = tid & 31;
    const int wid  = tid >> 5;
    const int half = (lane >> 4) & 1;
    const int r    = lane & 15;

    const int    b      = blockIdx.z;
    const int    h      = blockIdx.y;
    const int    qblock = blockIdx.x * 128;   // 8 waves * 16 rows
    const int    qw     = qblock + wid * 16;  // wave's first q row
    const size_t base   = ((size_t)b * SEQ) * D_MODEL + (size_t)h * DKH;

    // Q A-fragments for the whole d_k=128 (4 chunks of K=32), kept in VGPRs
    FragAB qf[4];
    {
        const u16* qrow = Q + base + (size_t)(qw + r) * D_MODEL;
        for (int kc = 0; kc < 4; ++kc) {
            qf[kc].q[0] = *(const v4u*)&qrow[kc * 32 + half * 8];
            qf[kc].q[1] = *(const v4u*)&qrow[kc * 32 + 16 + half * 8];
        }
    }

    v8f zero = {0.f, 0.f, 0.f, 0.f, 0.f, 0.f, 0.f, 0.f};
    v8f oacc[8];                        // 16 x 128 output accumulator
    for (int nc = 0; nc < 8; ++nc) oacc[nc] = zero;
    float mrow[8], lrow[8];             // per (half*8+i) row stats, lane-local
    for (int i = 0; i < 8; ++i) { mrow[i] = -1e30f; lrow[i] = 0.f; }

    const int ldrowk = tid >> 3;         // 0..31 (key within tile)
    const int ldc    = (tid & 7) * 16;   // d offset

    const int kend = qblock + 128;      // covers causal diagonal for every q in block
    for (int kt = 0; kt < kend; kt += KT) {
        // K tile: async global->LDS; V tile: load + transpose scatter (overlapped)
        {
            const u16* gk = &Km[base + (size_t)(kt + ldrowk) * D_MODEL + ldc];
            async_copy16(lds_off(&Ks[ldrowk][ldc]),     gk);
            async_copy16(lds_off(&Ks[ldrowk][ldc + 8]), gk + 8);
            const u16* gv = &Vm[base + (size_t)(kt + ldrowk) * D_MODEL + ldc];
            for (int e = 0; e < 16; ++e) Vt[ldc + e][ldrowk] = gv[e];
            S_WAIT_ASYNC_0;
        }
        __syncthreads();

        // S = Q @ K^T : two 16x16 score tiles (j over key halves)
        v8f s[2];
        for (int j = 0; j < 2; ++j) {
            s[j] = zero;
            for (int kc = 0; kc < 4; ++kc) {
                FragAB kf;                              // B col n=r -> key row j*16+r
                kf.q[0] = *(const v4u*)&Ks[j * 16 + r][kc * 32 + half * 8];
                kf.q[1] = *(const v4u*)&Ks[j * 16 + r][kc * 32 + 16 + half * 8];
                s[j] = __builtin_amdgcn_wmma_f32_16x16x32_bf16(
                    false, qf[kc].v, false, kf.v, (short)0, s[j], false, false);
            }
        }

        // causal mask
        for (int j = 0; j < 2; ++j)
            for (int i = 0; i < 8; ++i) {
                int qg = qw + half * 8 + i;
                int kg = kt + j * 16 + r;
                if (kg > qg) s[j][i] = -1e30f;
            }

        // online softmax: row reduce across the 16 lanes of this half
        float mnew[8], alpha[8];
        for (int i = 0; i < 8; ++i) {
            float t = fmaxf(s[0][i], s[1][i]);
            for (int msk = 1; msk <= 8; msk <<= 1) t = fmaxf(t, __shfl_xor(t, msk, 32));
            mnew[i]  = fmaxf(mrow[i], t);
            alpha[i] = __expf(mrow[i] - mnew[i]);
            mrow[i]  = mnew[i];
        }
        float rsum[8];
        for (int i = 0; i < 8; ++i) rsum[i] = 0.f;
        for (int j = 0; j < 2; ++j)
            for (int i = 0; i < 8; ++i) {
                float p = __expf(s[j][i] - mnew[i]);
                rsum[i] += p;
                Ps[wid][half * 8 + i][j * 16 + r] = f2bf(p);
            }
        for (int i = 0; i < 8; ++i) {
            float t = rsum[i];
            for (int msk = 1; msk <= 8; msk <<= 1) t += __shfl_xor(t, msk, 32);
            lrow[i] = lrow[i] * alpha[i] + t;
            for (int nc = 0; nc < 8; ++nc) oacc[nc][i] *= alpha[i];
        }

        // O += P @ V  (P from wave-private LDS as A-frag, V from transposed tile)
        FragAB pf;
        pf.q[0] = *(const v4u*)&Ps[wid][r][half * 8];
        pf.q[1] = *(const v4u*)&Ps[wid][r][16 + half * 8];
        for (int nc = 0; nc < 8; ++nc) {
            FragAB vf;                                   // B col n=r -> d = nc*16+r
            vf.q[0] = *(const v4u*)&Vt[nc * 16 + r][half * 8];
            vf.q[1] = *(const v4u*)&Vt[nc * 16 + r][16 + half * 8];
            oacc[nc] = __builtin_amdgcn_wmma_f32_16x16x32_bf16(
                false, pf.v, false, vf.v, (short)0, oacc[nc], false, false);
        }
        __syncthreads();
    }

    // epilogue: normalize and store bf16 in [B*S, H*d_k] layout
    for (int nc = 0; nc < 8; ++nc)
        for (int i = 0; i < 8; ++i) {
            int qg = qw + half * 8 + i;
            int d  = nc * 16 + r;
            O[base + (size_t)qg * D_MODEL + d] = f2bf(oacc[nc][i] / lrow[i]);
        }
}

// ---------------------------------------------------------------------------
// Host-side launch
// ---------------------------------------------------------------------------
extern "C" void kernel_launch(void* const* d_in, const int* in_sizes, int n_in,
                              void* d_out, int out_size, void* d_ws, size_t ws_size,
                              hipStream_t stream) {
    (void)in_sizes; (void)n_in; (void)out_size; (void)ws_size;
    const float* X  = (const float*)d_in[0];
    const float* Wq = (const float*)d_in[1];
    const float* Wk = (const float*)d_in[2];
    const float* Wv = (const float*)d_in[3];
    const float* Wo = (const float*)d_in[4];

    char*  ws  = (char*)d_ws;
    size_t off = 0;
    auto alloc = [&](size_t bytes) -> void* {
        void* p = ws + off;
        off += (bytes + 255) & ~(size_t)255;
        return p;
    };
    const size_t nX = (size_t)MTOT * D_MODEL;     // 8.39M
    const size_t nW = (size_t)D_MODEL * D_MODEL;  // 4.19M
    u16* Xb  = (u16*)alloc(nX * 2);
    u16* Wqb = (u16*)alloc(nW * 2);
    u16* Wkb = (u16*)alloc(nW * 2);
    u16* Wvb = (u16*)alloc(nW * 2);
    u16* Wob = (u16*)alloc(nW * 2);
    u16* Qb  = (u16*)alloc(nX * 2);
    u16* Kb  = (u16*)alloc(nX * 2);
    u16* Vb  = (u16*)alloc(nX * 2);
    u16* Ab  = (u16*)alloc(nX * 2);               // ~112 MB total workspace

    cvt_f32_bf16<<<1024, 256, 0, stream>>>(X,  Xb,  (int)nX);
    cvt_f32_bf16<<<512,  256, 0, stream>>>(Wq, Wqb, (int)nW);
    cvt_f32_bf16<<<512,  256, 0, stream>>>(Wk, Wkb, (int)nW);
    cvt_f32_bf16<<<512,  256, 0, stream>>>(Wv, Wvb, (int)nW);
    cvt_f32_bf16<<<512,  256, 0, stream>>>(Wo, Wob, (int)nW);

    dim3 gg(D_MODEL / 128, MTOT / 128);           // (16, 32)
    float qscale = 1.0f / sqrtf((float)DKH);
    gemm_xwT<true><<<gg, 256, 0, stream>>>(Xb, Wqb, Qb, MTOT, D_MODEL, D_MODEL, qscale);
    gemm_xwT<true><<<gg, 256, 0, stream>>>(Xb, Wkb, Kb, MTOT, D_MODEL, D_MODEL, 1.0f);
    gemm_xwT<true><<<gg, 256, 0, stream>>>(Xb, Wvb, Vb, MTOT, D_MODEL, D_MODEL, 1.0f);

    dim3 ga(SEQ / 128, HEADS, BATCH);             // (16, 16, 2)
    flash_attn<<<ga, 256, 0, stream>>>(Qb, Kb, Vb, Ab);

    gemm_xwT<false><<<gg, 256, 0, stream>>>(Ab, Wob, d_out, MTOT, D_MODEL, D_MODEL, 1.0f);
}
